// DecoderBlockBVL_66623532696057
// MI455X (gfx1250) — matrix operand order repaired and checked by reading, not compiled, not run on hardware
//
#include <hip/hip_runtime.h>
#include <cstdint>
#include <cstddef>

// ---------------------------------------------------------------------------
// Types (plain ext-vectors so unions stay trivially constructible)
// ---------------------------------------------------------------------------
typedef __bf16 bf16;
typedef bf16         bf16x16 __attribute__((ext_vector_type(16)));
typedef float        floatx8 __attribute__((ext_vector_type(8)));
typedef uint32_t     u32x4   __attribute__((ext_vector_type(4)));
typedef uint32_t     u32x2   __attribute__((ext_vector_type(2)));
typedef float        f32x4   __attribute__((ext_vector_type(4)));
typedef unsigned int uu32x4  __attribute__((ext_vector_type(4)));
typedef int          ii32x4  __attribute__((ext_vector_type(4)));
typedef int          ii32x8  __attribute__((ext_vector_type(8)));

#define BB   2
#define VV   8
#define LL   256
#define CC   768
#define HH   12
#define HD   64
#define SS   (VV * LL)   /* 2048 */
#define MM   (BB * SS)   /* 4096 rows everywhere */
#define HID  3072

// ---------------------------------------------------------------------------
// Tensor Data Mover support (gfx1250 async tensor -> LDS), probe-guarded.
// ---------------------------------------------------------------------------
#if defined(__has_builtin)
#if __has_builtin(__builtin_amdgcn_tensor_load_to_lds)
#define HAVE_TDM 1
#endif
#endif
#ifndef HAVE_TDM
#define HAVE_TDM 0
#endif

#if HAVE_TDM
// 2D tile load: tileRows x tileCols (bf16) from row-major tensor with row
// length tensorRowLen (elements), into LDS at lds_byte_off.  TDM pad feature
// inserts 4 DWORDs after every 16 DWORDs -> padded LDS row stride of
// 32 + 8 = 40 halves (matches the fragment loaders' LDSS).
__device__ __forceinline__ void tdm_load_tile(uint32_t lds_byte_off,
                                              const bf16* gptr, int tileRows,
                                              int tileCols, int tensorRowLen,
                                              int tensorRows) {
  const uint64_t ga = (uint64_t)(uintptr_t)gptr;
  uu32x4 g0;
  g0[0] = 1u;                                   // count=1 (valid user D#)
  g0[1] = lds_byte_off;                         // D#.lds_addr (bytes)
  g0[2] = (uint32_t)ga;                         // global_addr[31:0]
  g0[3] = (uint32_t)((ga >> 32) & 0x01FFFFFFu)  // global_addr[56:32]
          | (2u << 30);                         // type = 2 ("image")
  ii32x8 g1;
  // data_size=1 (2B) | pad_enable | pad_interval=3 (16 DW) | pad_amount=3 (4 DW)
  g1[0] = (1 << 16) | (1 << 20) | (3 << 22) | (3 << 25);
  g1[1] = (int)((uint32_t)tensorRowLen << 16);                 // tensor_dim0 lo16
  g1[2] = (int)(((uint32_t)tensorRowLen >> 16) |
                ((uint32_t)tensorRows << 16));                 // dim0 hi | dim1 lo
  g1[3] = (int)(((uint32_t)tensorRows >> 16) |
                ((uint32_t)tileCols << 16));                   // dim1 hi | tile_dim0
  g1[4] = tileRows;                                            // tile_dim1 (dim2=0)
  g1[5] = tensorRowLen;                                        // dim0 stride lo32
  g1[6] = 0;                                                   // stride hi | dim1 stride lo
  g1[7] = 0;
  ii32x4 gz = {0, 0, 0, 0};
#if __clang_major__ >= 23
  ii32x8 gz8 = {0, 0, 0, 0, 0, 0, 0, 0};
  __builtin_amdgcn_tensor_load_to_lds(g0, g1, gz, gz, gz8, 0);
#else
  __builtin_amdgcn_tensor_load_to_lds(g0, g1, gz, gz, 0);
#endif
}
#endif  // HAVE_TDM

// ---------------------------------------------------------------------------
// WMMA helper (CDNA5 gfx1250, wave32): D = A(16x32 bf16) * B(32x16 bf16) + C(f32)
// ---------------------------------------------------------------------------
__device__ __forceinline__ floatx8 wmma_bf16(bf16x16 a, bf16x16 b, floatx8 c) {
  return __builtin_amdgcn_wmma_f32_16x16x32_bf16(false, a, false, b, (short)0, c,
                                                 false, false);
}

union Frag { bf16x16 v; u32x4 q[2]; };

// A-fragment (16x32, 16-bit).  lane%16 = M row; halves i hold
// K = 16*g + i (contiguous, matching the B loaders).
__device__ __forceinline__ bf16x16 load_frag_a(const bf16* base, int stride,
                                               int lane16, int g) {
  const bf16* p = base + (size_t)lane16 * stride + g * 16;
  Frag f;
  f.q[0] = *(const u32x4*)(p);
  f.q[1] = *(const u32x4*)(p + 8);
  return f.v;
}

// B-fragment (32x16, 16-bit) gathered from a row-major Mat[n][k]:
// lane%16 = N; halves i hold K = 16*g + i (contiguous 32B per lane).
__device__ __forceinline__ bf16x16 load_frag_b(const bf16* base, int stride,
                                               int lane16, int g) {
  const bf16* p = base + (size_t)lane16 * stride + g * 16;
  Frag f;
  f.q[0] = *(const u32x4*)(p);
  f.q[1] = *(const u32x4*)(p + 8);
  return f.v;
}

// B-fragment from row-major V[k][n] (keys x dims): lane%16 = N column (dim),
// halves i walk key rows 16*g + i (strided).
__device__ __forceinline__ bf16x16 load_frag_v(const bf16* base, int stride,
                                               int lane16, int g) {
  bf16x16 f;
  const bf16* p = base + lane16 + (size_t)(16 * g) * stride;
#pragma unroll
  for (int i = 0; i < 16; ++i) f[i] = p[(size_t)i * stride];
  return f;
}

// ---------------------------------------------------------------------------
// One-shot weight conversion f32 -> bf16 (hoists cvt + halves weight traffic)
// ---------------------------------------------------------------------------
__global__ __launch_bounds__(256) void cvt_w_bf16(const float* __restrict__ in,
                                                  bf16* __restrict__ out) {
  const int i = (blockIdx.x * 256 + threadIdx.x) * 4;
  f32x4 v = *(const f32x4*)(in + i);
  union { bf16 h[4]; u32x2 q; } cv;
#pragma unroll
  for (int j = 0; j < 4; ++j) cv.h[j] = (bf16)v[j];
  *(u32x2*)(out + i) = cv.q;
}

// ---------------------------------------------------------------------------
// LayerNorm: f32 [rows, 768] -> bf16 [rows, 768]
// ---------------------------------------------------------------------------
__global__ __launch_bounds__(256) void ln_bf16(const float* __restrict__ X,
                                               const float* __restrict__ gw,
                                               const float* __restrict__ bw,
                                               bf16* __restrict__ out) {
  const int row = blockIdx.x;
  const int tid = threadIdx.x;
  const float* xr = X + (size_t)row * CC;
  float vals[3];
  float s = 0.f, ss = 0.f;
#pragma unroll
  for (int i = 0; i < 3; ++i) {
    float v = xr[tid + i * 256];
    vals[i] = v; s += v; ss += v * v;
  }
#pragma unroll
  for (int off = 16; off >= 1; off >>= 1) {
    s  += __shfl_xor(s,  off, 32);
    ss += __shfl_xor(ss, off, 32);
  }
  __shared__ float rs[8], rss[8];
  const int wave = tid >> 5, lane = tid & 31;
  if (lane == 0) { rs[wave] = s; rss[wave] = ss; }
  __syncthreads();
  s = 0.f; ss = 0.f;
#pragma unroll
  for (int i = 0; i < 8; ++i) { s += rs[i]; ss += rss[i]; }
  const float mean = s * (1.f / CC);
  const float var  = ss * (1.f / CC) - mean * mean;
  const float inv  = rsqrtf(var + 1e-5f);
  bf16* orow = out + (size_t)row * CC;
#pragma unroll
  for (int i = 0; i < 3; ++i) {
    int c = tid + i * 256;
    orow[c] = (bf16)((vals[i] - mean) * inv * gw[c] + bw[c]);
  }
}

// ---------------------------------------------------------------------------
// GEMM: Y[M,N] = act(A_bf16[M,K] @ W_bf16[N,K]^T + bias) (+ residual_f32)
// Block tile 128x128, K-step 32, double-buffered LDS.  TDM (tensor_load_to_lds)
// streams tile k+1 while 8 waves (4Mx2N, 32x64 each) run WMMAs on tile k.
// ---------------------------------------------------------------------------
#define GT_M 128
#define GT_N 128
#define GT_K 32
#define LDSS 40  // padded LDS row stride in halves (80B, 16B-aligned)

template <bool BIAS, bool RES, bool GELU, bool OUTBF>
__global__ __launch_bounds__(256) void gemm_wmma(
    const bf16* __restrict__ A, const bf16* __restrict__ W,
    const float* __restrict__ bias, const float* __restrict__ resid,
    bf16* __restrict__ outB, float* __restrict__ outF, int Mdim, int Ndim,
    int Kdim) {
  __shared__ bf16 As[2][GT_M * LDSS];
  __shared__ bf16 Bs[2][GT_N * LDSS];
  const int tid = threadIdx.x;
  const int lane = tid & 31, wave = tid >> 5;
  const int lane16 = lane & 15, g = lane >> 4;
  const int wm = wave & 3, wn = wave >> 2;
  const int mBase = blockIdx.y * GT_M;
  const int nBase = blockIdx.x * GT_N;

  const floatx8 fz = {0.f, 0.f, 0.f, 0.f, 0.f, 0.f, 0.f, 0.f};
  floatx8 acc[2][4];
#pragma unroll
  for (int s2 = 0; s2 < 2; ++s2)
#pragma unroll
    for (int j = 0; j < 4; ++j) acc[s2][j] = fz;

  const bf16* Abase = A + (size_t)mBase * Kdim;
  const bf16* Wbase = W + (size_t)nBase * Kdim;
  const int nk = Kdim / GT_K;

#if HAVE_TDM
  if (wave == 0) {  // prologue: TDM fills buffer 0 (EXEC-independent, per-wave)
    tdm_load_tile((uint32_t)(uintptr_t)&As[0][0], Abase, GT_M, GT_K, Kdim, GT_M);
    tdm_load_tile((uint32_t)(uintptr_t)&Bs[0][0], Wbase, GT_N, GT_K, Kdim, GT_N);
  }
#endif

  for (int ki = 0; ki < nk; ++ki) {
    const int buf = ki & 1;
#if HAVE_TDM
    if (wave == 0) {
      if (ki + 1 < nk) {
        // stream tile k+1 into the other buffer, then wait only for tile k
        tdm_load_tile((uint32_t)(uintptr_t)&As[buf ^ 1][0],
                      Abase + (size_t)(ki + 1) * GT_K, GT_M, GT_K, Kdim, GT_M);
        tdm_load_tile((uint32_t)(uintptr_t)&Bs[buf ^ 1][0],
                      Wbase + (size_t)(ki + 1) * GT_K, GT_N, GT_K, Kdim, GT_N);
        __builtin_amdgcn_s_wait_tensorcnt(2);
      } else {
        __builtin_amdgcn_s_wait_tensorcnt(0);
      }
    }
#else
    // fallback: cooperative vector-load fill (32B per thread per tile)
    {
      const int rowL = tid >> 1;
      const int part = tid & 1;
      const bf16* ag = Abase + (size_t)rowL * Kdim + ki * GT_K + part * 16;
      const bf16* wg = Wbase + (size_t)rowL * Kdim + ki * GT_K + part * 16;
      u32x4 a0 = *(const u32x4*)(ag);
      u32x4 a1 = *(const u32x4*)(ag + 8);
      u32x4 b0 = *(const u32x4*)(wg);
      u32x4 b1 = *(const u32x4*)(wg + 8);
      bf16* asd = &As[buf][rowL * LDSS + part * 16];
      *(u32x4*)(asd)     = a0;
      *(u32x4*)(asd + 8) = a1;
      bf16* bsd = &Bs[buf][rowL * LDSS + part * 16];
      *(u32x4*)(bsd)     = b0;
      *(u32x4*)(bsd + 8) = b1;
    }
#endif
    __syncthreads();  // publish tile ki to all waves

    bf16x16 af[2], bfm[4];
#pragma unroll
    for (int s2 = 0; s2 < 2; ++s2)
      af[s2] =
          load_frag_a(&As[buf][(wm * 32 + s2 * 16) * LDSS], LDSS, lane16, g);
#pragma unroll
    for (int j = 0; j < 4; ++j)
      bfm[j] =
          load_frag_b(&Bs[buf][(wn * 64 + j * 16) * LDSS], LDSS, lane16, g);
#pragma unroll
    for (int s2 = 0; s2 < 2; ++s2)
#pragma unroll
      for (int j = 0; j < 4; ++j)
        acc[s2][j] = wmma_bf16(af[s2], bfm[j], acc[s2][j]);

    __syncthreads();  // buffer `buf` free before TDM refills it next round
  }

  // --- epilogue: C layout => lane holds (m = r + 8*g, n = lane16) per VGPR r
#pragma unroll
  for (int s2 = 0; s2 < 2; ++s2)
#pragma unroll
    for (int j = 0; j < 4; ++j) {
      const int n = nBase + wn * 64 + j * 16 + lane16;
      const float bv = BIAS ? bias[n] : 0.f;
#pragma unroll
      for (int r = 0; r < 8; ++r) {
        const int m = mBase + wm * 32 + s2 * 16 + r + 8 * g;
        const size_t off = (size_t)m * Ndim + n;
        float v = acc[s2][j][r] + bv;
        if (GELU) v = 0.5f * v * (1.f + erff(v * 0.70710678118654752f));
        if (RES) v += resid[off];
        if (OUTBF) outB[off] = (bf16)v;
        else       outF[off] = v;
      }
    }
  (void)Mdim;
}

// ---------------------------------------------------------------------------
// Flash attention (wave32 WMMA): one wave per 16-query tile, 32-key blocks.
// Block-causal mask is tile-uniform -> reduces to a key-range bound.
// ---------------------------------------------------------------------------
template <bool CAUSAL>
__global__ __launch_bounds__(256) void attn_wmma(const bf16* __restrict__ Qp,
                                                 const bf16* __restrict__ Kp,
                                                 const bf16* __restrict__ Vp,
                                                 bf16* __restrict__ Op,
                                                 int seqLen, int stride) {
  __shared__ bf16 pstage[8][512];  // per-wave 16x32 P tile (row stride 32)
  const int tid = threadIdx.x, lane = tid & 31, wave = tid >> 5;
  const int lane16 = lane & 15, g = lane >> 4;
  const int chunks = seqLen >> 7;           // 128 queries per block
  const int ch  = blockIdx.x % chunks;
  const int t   = blockIdx.x / chunks;
  const int h   = t % HH;
  const int seq = t / HH;
  const int rowBase = seq * seqLen;
  const int q0 = ch * 128 + wave * 16;
  const int colQ = h * HD;

  const bf16* qrow = Qp + (size_t)(rowBase + q0) * stride + colQ;
  const bf16x16 qf0 = load_frag_a(qrow, stride, lane16, g);       // d 0..31
  const bf16x16 qf1 = load_frag_a(qrow + 32, stride, lane16, g);  // d 32..63

  const floatx8 fz = {0.f, 0.f, 0.f, 0.f, 0.f, 0.f, 0.f, 0.f};
  floatx8 o0 = fz, o1 = fz, o2 = fz, o3 = fz;
  float mrow[8], lrow[8];
#pragma unroll
  for (int r = 0; r < 8; ++r) { mrow[r] = -3.0e38f; lrow[r] = 0.f; }

  int kmax;
  if (CAUSAL) {
    const int vq = q0 / LL;
    const int vm = vq < 1 ? 1 : vq;
    kmax = (vm + 1) * LL;   // views {0,1} see {0,1}; view v>=2 sees [0..v]
  } else {
    kmax = seqLen;
  }
  const float scale = 0.125f;  // 64^-0.5

  for (int kb = 0; kb < kmax; kb += 32) {
    // ---- S = Q . K^T for two 16-key sub-tiles
    floatx8 s0 = fz, s1 = fz;
    const bf16* kr0 = Kp + (size_t)(rowBase + kb) * stride + colQ;
    const bf16* kr1 = Kp + (size_t)(rowBase + kb + 16) * stride + colQ;
    s0 = wmma_bf16(qf0, load_frag_b(kr0, stride, lane16, g), s0);
    s1 = wmma_bf16(qf0, load_frag_b(kr1, stride, lane16, g), s1);
    s0 = wmma_bf16(qf1, load_frag_b(kr0 + 32, stride, lane16, g), s0);
    s1 = wmma_bf16(qf1, load_frag_b(kr1 + 32, stride, lane16, g), s1);

    // ---- online softmax (row reductions across each 16-lane half)
    float alpha[8];
#pragma unroll
    for (int r = 0; r < 8; ++r) {
      const float a = s0[r] * scale;
      const float b = s1[r] * scale;
      float mx = fmaxf(a, b);
#pragma unroll
      for (int off = 8; off >= 1; off >>= 1)
        mx = fmaxf(mx, __shfl_xor(mx, off, 32));
      const float nm = fmaxf(mrow[r], mx);
      const float al = __expf(mrow[r] - nm);
      mrow[r] = nm;
      const float p0 = __expf(a - nm);
      const float p1 = __expf(b - nm);
      float rsum = p0 + p1;
#pragma unroll
      for (int off = 8; off >= 1; off >>= 1)
        rsum += __shfl_xor(rsum, off, 32);
      lrow[r] = lrow[r] * al + rsum;
      alpha[r] = al;
      s0[r] = p0;
      s1[r] = p1;
    }
#pragma unroll
    for (int r = 0; r < 8; ++r) {
      o0[r] *= alpha[r]; o1[r] *= alpha[r];
      o2[r] *= alpha[r]; o3[r] *= alpha[r];
    }

    // ---- transpose P (C layout) -> A layout via per-wave LDS tile
    bf16* pb = &pstage[wave][0];
#pragma unroll
    for (int r = 0; r < 8; ++r) {
      const int m = r + 8 * g;
      pb[m * 32 + lane16]      = (bf16)s0[r];
      pb[m * 32 + 16 + lane16] = (bf16)s1[r];
    }
    const bf16x16 pf = load_frag_a(pb, 32, lane16, g);

    // ---- O += P . V (four 16-wide d segments)
    const bf16* vr = Vp + (size_t)(rowBase + kb) * stride + colQ;
    o0 = wmma_bf16(pf, load_frag_v(vr + 0,  stride, lane16, g), o0);
    o1 = wmma_bf16(pf, load_frag_v(vr + 16, stride, lane16, g), o1);
    o2 = wmma_bf16(pf, load_frag_v(vr + 32, stride, lane16, g), o2);
    o3 = wmma_bf16(pf, load_frag_v(vr + 48, stride, lane16, g), o3);
  }

  // ---- normalize and write O as bf16 rows into [M, 768]
#pragma unroll
  for (int r = 0; r < 8; ++r) {
    const int m = rowBase + q0 + r + 8 * g;
    const float invl = 1.f / lrow[r];
    const size_t off = (size_t)m * CC + colQ + lane16;
    Op[off + 0]  = (bf16)(o0[r] * invl);
    Op[off + 16] = (bf16)(o1[r] * invl);
    Op[off + 32] = (bf16)(o2[r] * invl);
    Op[off + 48] = (bf16)(o3[r] * invl);
  }
}

// ---------------------------------------------------------------------------
// Host-side orchestration
// ---------------------------------------------------------------------------
extern "C" void kernel_launch(void* const* d_in, const int* in_sizes, int n_in,
                              void* d_out, int out_size, void* d_ws,
                              size_t ws_size, hipStream_t stream) {
  (void)in_sizes; (void)n_in; (void)out_size; (void)ws_size;
  const float* x           = (const float*)d_in[0];
  const float* ln1_g       = (const float*)d_in[1];
  const float* ln1_b       = (const float*)d_in[2];
  const float* ln2_g       = (const float*)d_in[3];
  const float* ln2_b       = (const float*)d_in[4];
  const float* ln3_g       = (const float*)d_in[5];
  const float* ln3_b       = (const float*)d_in[6];
  const float* qkv_w       = (const float*)d_in[7];
  const float* attn_proj_w = (const float*)d_in[8];
  const float* attn_proj_b = (const float*)d_in[9];
  const float* q_w         = (const float*)d_in[10];
  const float* k_w         = (const float*)d_in[11];
  const float* v_w         = (const float*)d_in[12];
  const float* cproj_w     = (const float*)d_in[13];
  const float* cproj_b     = (const float*)d_in[14];
  const float* fc1_w       = (const float*)d_in[15];
  const float* fc1_b       = (const float*)d_in[16];
  const float* fc2_w       = (const float*)d_in[17];
  const float* fc2_b       = (const float*)d_in[18];
  float* out = (float*)d_out;

  char* ws = (char*)d_ws;
  size_t off = 0;
  bf16* hbf   = (bf16*)(ws + off); off += (size_t)MM * CC * sizeof(bf16);
  bf16* qkvb  = (bf16*)(ws + off); off += (size_t)MM * 3 * CC * sizeof(bf16);
  bf16* attno = (bf16*)(ws + off); off += (size_t)MM * CC * sizeof(bf16);
  bf16* fc1o  = (bf16*)(ws + off); off += (size_t)MM * HID * sizeof(bf16);
  // bf16 weight mirrors
  bf16* wqkv  = (bf16*)(ws + off); off += (size_t)3 * CC * CC * sizeof(bf16);
  bf16* wap   = (bf16*)(ws + off); off += (size_t)CC * CC * sizeof(bf16);
  bf16* wq    = (bf16*)(ws + off); off += (size_t)CC * CC * sizeof(bf16);
  bf16* wk    = (bf16*)(ws + off); off += (size_t)CC * CC * sizeof(bf16);
  bf16* wv    = (bf16*)(ws + off); off += (size_t)CC * CC * sizeof(bf16);
  bf16* wcp   = (bf16*)(ws + off); off += (size_t)CC * CC * sizeof(bf16);
  bf16* wfc1  = (bf16*)(ws + off); off += (size_t)HID * CC * sizeof(bf16);
  bf16* wfc2  = (bf16*)(ws + off); off += (size_t)CC * HID * sizeof(bf16);

  bf16* qb2 = qkvb;
  bf16* kb2 = qkvb + (size_t)MM * CC;
  bf16* vb2 = qkvb + (size_t)2 * MM * CC;

  const dim3 blk(256);

  // ---- one-shot weight conversion to bf16 (4 elems/thread) --------------
  cvt_w_bf16<<<(3 * CC * CC) / 1024, blk, 0, stream>>>(qkv_w, wqkv);
  cvt_w_bf16<<<(CC * CC) / 1024, blk, 0, stream>>>(attn_proj_w, wap);
  cvt_w_bf16<<<(CC * CC) / 1024, blk, 0, stream>>>(q_w, wq);
  cvt_w_bf16<<<(CC * CC) / 1024, blk, 0, stream>>>(k_w, wk);
  cvt_w_bf16<<<(CC * CC) / 1024, blk, 0, stream>>>(v_w, wv);
  cvt_w_bf16<<<(CC * CC) / 1024, blk, 0, stream>>>(cproj_w, wcp);
  cvt_w_bf16<<<(HID * CC) / 1024, blk, 0, stream>>>(fc1_w, wfc1);
  cvt_w_bf16<<<(CC * HID) / 1024, blk, 0, stream>>>(fc2_w, wfc2);

  // ---- stage 1: per-view self-attention -------------------------------
  ln_bf16<<<MM, blk, 0, stream>>>(x, ln1_g, ln1_b, hbf);
  gemm_wmma<false, false, false, true>
      <<<dim3(3 * CC / 128, MM / 128), blk, 0, stream>>>(
          hbf, wqkv, nullptr, nullptr, qkvb, nullptr, MM, 3 * CC, CC);
  attn_wmma<false><<<(BB * VV) * HH * (LL / 128), blk, 0, stream>>>(
      qkvb, qkvb + CC, qkvb + 2 * CC, attno, LL, 3 * CC);
  gemm_wmma<true, true, false, false>
      <<<dim3(CC / 128, MM / 128), blk, 0, stream>>>(
          attno, wap, attn_proj_b, x, nullptr, out, MM, CC, CC);

  // ---- stage 2: block-causal cross-view attention ---------------------
  ln_bf16<<<MM, blk, 0, stream>>>(out, ln2_g, ln2_b, hbf);
  gemm_wmma<false, false, false, true>
      <<<dim3(CC / 128, MM / 128), blk, 0, stream>>>(
          hbf, wq, nullptr, nullptr, qb2, nullptr, MM, CC, CC);
  gemm_wmma<false, false, false, true>
      <<<dim3(CC / 128, MM / 128), blk, 0, stream>>>(
          hbf, wk, nullptr, nullptr, kb2, nullptr, MM, CC, CC);
  gemm_wmma<false, false, false, true>
      <<<dim3(CC / 128, MM / 128), blk, 0, stream>>>(
          hbf, wv, nullptr, nullptr, vb2, nullptr, MM, CC, CC);
  attn_wmma<true><<<BB * HH * (SS / 128), blk, 0, stream>>>(
      qb2, kb2, vb2, attno, SS, CC);
  gemm_wmma<true, true, false, false>
      <<<dim3(CC / 128, MM / 128), blk, 0, stream>>>(
          attno, wcp, cproj_b, out, nullptr, out, MM, CC, CC);

  // ---- stage 3: MLP with exact GELU -----------------------------------
  ln_bf16<<<MM, blk, 0, stream>>>(out, ln3_g, ln3_b, hbf);
  gemm_wmma<true, false, true, true>
      <<<dim3(HID / 128, MM / 128), blk, 0, stream>>>(
          hbf, wfc1, fc1_b, nullptr, fc1o, nullptr, MM, HID, CC);
  gemm_wmma<true, true, false, false>
      <<<dim3(CC / 128, MM / 128), blk, 0, stream>>>(
          fc1o, wfc2, fc2_b, out, nullptr, out, MM, CC, HID);
}
